// MaxViTBlock_343597384425
// MI455X (gfx1250) — matrix operand order repaired and checked by reading, not compiled, not run on hardware
//
#include <hip/hip_runtime.h>
#include <math.h>

typedef _Float16 h16;
typedef __attribute__((ext_vector_type(16))) _Float16 v16h;
typedef __attribute__((ext_vector_type(8)))  _Float16 v8h;
typedef __attribute__((ext_vector_type(8)))  float    v8f;

__device__ __forceinline__ float gelu_exact(float x){
  return 0.5f * x * (1.0f + erff(x * 0.70710678118654752f));
}
__device__ __forceinline__ float sigmf(float x){
  return 1.0f / (1.0f + __expf(-x));
}

// ---- WMMA fragment helpers (wave32, 16x16x32 f16) ----------------------------
// A layout in memory: [M][K], K contiguous, row stride rs (halves, mult of 8)
__device__ __forceinline__ v16h frag_a(const h16* p, int rs, int lane){
  int m = lane & 15, g = lane >> 4;
  const h16* r = p + m * rs + g * 8;
  union { v16h v; v8h h[2]; } u;
  u.h[0] = *(const v8h*)(r);        // K = k0+g*8 .. +7
  u.h[1] = *(const v8h*)(r + 16);   // K = k0+16+g*8 .. +7
  return u.v;
}
// B layout in memory: [N][K], K contiguous, row stride rs (halves, mult of 8)
__device__ __forceinline__ v16h frag_b(const h16* p, int rs, int lane){
  int n = lane & 15, g = lane >> 4;
  const h16* r = p + n * rs + g * 8;
  union { v16h v; v8h h[2]; } u;
  u.h[0] = *(const v8h*)(r);        // K = k0+g*8 .. +7
  u.h[1] = *(const v8h*)(r + 16);   // K = k0+16+g*8 .. +7
  return u.v;
}
__device__ __forceinline__ v8f wmma16(v16h a, v16h b, v8f c){
  return __builtin_amdgcn_wmma_f32_16x16x32_f16(false, a, false, b, (short)0, c, false, false);
}

// ---- K0: f32 -> f16 transpose copy (weights into fragment-friendly layout) ---
__global__ void k_cvt_t(const float* __restrict__ src, h16* __restrict__ dst,
                        int rows, int cols){
  int i = blockIdx.x * 256 + threadIdx.x;
  if (i < rows * cols){
    int r = i / cols, c = i - r * cols;
    dst[c * rows + r] = (h16)src[i];
  }
}

// ---- K1: conv1 (1x1, 64->512) + bias + BN1 + GELU, out f16 -------------------
// GEMM: M=512(d) N=50176(pos) K=64(c), per batch.
__global__ void k_conv1(const float* __restrict__ x, const h16* __restrict__ wT,
                        const float* __restrict__ bias, const float* __restrict__ g,
                        const float* __restrict__ bb, h16* __restrict__ h1){
  const int HW = 224 * 224;
  int b  = blockIdx.z;
  int n0 = blockIdx.x * 128;
  int m0 = blockIdx.y * 128;
  __shared__ __align__(16) h16 Bs[128 * 72];
  __shared__ float pg[128], pb[128], pbias[128];
  int tid = threadIdx.x;
  for (int i = tid; i < 128 * 64; i += 256){
    int n = i & 127, k = i >> 7;
    Bs[n * 72 + k] = (h16)x[((size_t)(b * 64 + k)) * HW + n0 + n];
  }
  if (tid < 128){ pg[tid] = g[m0 + tid]; pb[tid] = bb[m0 + tid]; pbias[tid] = bias[m0 + tid]; }
  __syncthreads();
  int wave = tid >> 5, lane = tid & 31;
  const h16* aBase = wT + (size_t)(m0 + wave * 16) * 64;
  v16h a0 = frag_a(aBase,      64, lane);
  v16h a1 = frag_a(aBase + 32, 64, lane);
  int gsel = lane >> 4;
  #pragma unroll
  for (int nt = 0; nt < 8; ++nt){
    v8f acc = {};
    v16h b0 = frag_b(&Bs[(nt * 16) * 72],      72, lane);
    v16h b1 = frag_b(&Bs[(nt * 16) * 72 + 32], 72, lane);
    acc = wmma16(a0, b0, acc);
    acc = wmma16(a1, b1, acc);
    int n = n0 + nt * 16 + (lane & 15);
    #pragma unroll
    for (int r = 0; r < 8; ++r){
      int ml = wave * 16 + r + 8 * gsel;   // local m in [0,128)
      float v = gelu_exact((acc[r] + pbias[ml]) * pg[ml] + pb[ml]);
      h1[((size_t)(b * 512 + m0 + ml)) * HW + n] = (h16)v;
    }
  }
}

// ---- K2: depthwise 3x3 stride2 pad1 + bias + BN2 + GELU, out f16 -------------
__global__ void k_dwconv(const h16* __restrict__ h1, const float* __restrict__ w,
                         const float* __restrict__ bias, const float* __restrict__ g,
                         const float* __restrict__ bb, h16* __restrict__ h2){
  int b = blockIdx.z, c = blockIdx.y;
  int p0 = blockIdx.x * 256 + threadIdx.x;        // 0..12543
  int oh = p0 / 112, ow = p0 - oh * 112;
  float acc = bias[c];
  const h16* in = h1 + ((size_t)(b * 512 + c)) * (224 * 224);
  #pragma unroll
  for (int kh = 0; kh < 3; ++kh){
    int ih = 2 * oh - 1 + kh;
    if ((unsigned)ih < 224u){
      #pragma unroll
      for (int kw = 0; kw < 3; ++kw){
        int iw = 2 * ow - 1 + kw;
        if ((unsigned)iw < 224u)
          acc += (float)in[ih * 224 + iw] * w[c * 9 + kh * 3 + kw];
      }
    }
  }
  float v = gelu_exact(acc * g[c] + bb[c]);
  h2[((size_t)(b * 512 + c)) * 12544 + p0] = (h16)v;
}

// ---- K3a: deterministic per-(b,c) plane sum for SE ---------------------------
__global__ void k_sesum(const h16* __restrict__ h2, float* __restrict__ sums){
  int c = blockIdx.x, b = blockIdx.y;
  const h16* p = h2 + ((size_t)(b * 512 + c)) * 12544;
  float a = 0.f;
  for (int i = threadIdx.x; i < 12544; i += 256) a += (float)p[i];
  __shared__ float red[256];
  red[threadIdx.x] = a; __syncthreads();
  for (int s = 128; s > 0; s >>= 1){
    if (threadIdx.x < s) red[threadIdx.x] += red[threadIdx.x + s];
    __syncthreads();
  }
  if (threadIdx.x == 0) sums[b * 512 + c] = red[0];
}

// ---- K3b: SE MLP: s = sigmoid(silu(mean@w1)@w2) ------------------------------
__global__ void k_se(const float* __restrict__ sums, const float* __restrict__ w1,
                     const float* __restrict__ w2, float* __restrict__ s){
  int b = blockIdx.x, t = threadIdx.x;     // 512 threads
  __shared__ float mean[512];
  __shared__ float hid[128];
  mean[t] = sums[b * 512 + t] * (1.0f / 12544.0f);
  __syncthreads();
  if (t < 128){
    float a = 0.f;
    for (int c = 0; c < 512; ++c) a += mean[c] * w1[c * 128 + t];
    hid[t] = a * sigmf(a);                  // silu
  }
  __syncthreads();
  float a = 0.f;
  for (int j = 0; j < 128; ++j) a += hid[j] * w2[j * 512 + t];
  s[b * 512 + t] = sigmf(a);
}

// ---- K4: (h2 * s) conv2 (1x1 512->128) + bias + BN3, out f32 -----------------
// GEMM: M=128(dout) N=12544(pos) K=512(c) per batch.
__global__ void k_conv2(const h16* __restrict__ h2, const float* __restrict__ s,
                        const h16* __restrict__ wT, const float* __restrict__ bias,
                        const float* __restrict__ g, const float* __restrict__ bb,
                        float* __restrict__ x0){
  const int HW = 12544;
  int b = blockIdx.z;
  int n0 = blockIdx.x * 128;
  __shared__ __align__(16) h16 Bs[128 * 72];
  __shared__ float sc[512];
  int tid = threadIdx.x, wave = tid >> 5, lane = tid & 31;
  for (int i = tid; i < 512; i += 256) sc[i] = s[b * 512 + i];
  v8f acc[8];
  #pragma unroll
  for (int i = 0; i < 8; ++i){ v8f z = {}; acc[i] = z; }
  int mt = wave * 16;
  for (int kc = 0; kc < 8; ++kc){
    __syncthreads();
    for (int i = tid; i < 128 * 64; i += 256){
      int n = i & 127, k = i >> 7;
      int c = kc * 64 + k;
      Bs[n * 72 + k] = (h16)((float)h2[((size_t)(b * 512 + c)) * HW + n0 + n] * sc[c]);
    }
    __syncthreads();
    const h16* aB = wT + (size_t)mt * 512 + kc * 64;
    v16h a0 = frag_a(aB,      512, lane);
    v16h a1 = frag_a(aB + 32, 512, lane);
    #pragma unroll
    for (int nt = 0; nt < 8; ++nt){
      v16h b0 = frag_b(&Bs[nt * 16 * 72],      72, lane);
      v16h b1 = frag_b(&Bs[nt * 16 * 72 + 32], 72, lane);
      acc[nt] = wmma16(a0, b0, acc[nt]);
      acc[nt] = wmma16(a1, b1, acc[nt]);
    }
  }
  int gsel = lane >> 4;
  #pragma unroll
  for (int nt = 0; nt < 8; ++nt){
    int n = n0 + nt * 16 + (lane & 15);
    #pragma unroll
    for (int r = 0; r < 8; ++r){
      int m = mt + r + 8 * gsel;
      x0[((size_t)(b * 128 + m)) * HW + n] = (acc[nt][r] + bias[m]) * g[m] + bb[m];
    }
  }
}

// ---- K5: windowed attention (block mode=0 / grid mode=1), whole window in LDS
// LDS carve (halves): LNX[64][136], Q[64][136], K[64][136], VT[128][72],
//                     ATT[64][72], OUT[64][136], then SIM f32 [64][68]
__global__ void k_attn(const float* __restrict__ xin, float* __restrict__ xout,
                       const float* __restrict__ ln_g, const float* __restrict__ ln_b,
                       const h16* __restrict__ qkvT, const h16* __restrict__ outT,
                       const float* __restrict__ rel, int mode){
  extern __shared__ h16 lds[];
  h16* LNX = lds;                 // 8704
  h16* Q   = LNX + 64 * 136;      // 8704
  h16* Kk  = Q   + 64 * 136;      // 8704
  h16* VT  = Kk  + 64 * 136;      // 9216
  h16* ATT = VT  + 128 * 72;      // 4608
  h16* OUT = ATT + 64 * 72;       // 8704
  float* SIM = (float*)(OUT + 64 * 136);  // 64*68 f32
  const int HW = 12544;
  int gx = blockIdx.x, gy = blockIdx.y, b = blockIdx.z;
  int tid = threadIdx.x, wave = tid >> 5, lane = tid & 31;

  // Phase 1: load tokens, LayerNorm, write f16 [t][d] (pad rows zeroed)
  for (int t = wave; t < 64; t += 8){
    if (t < 49){
      int w1 = t / 7, w2 = t - w1 * 7;
      int hh = mode ? (w1 * 16 + gx) : (gx * 7 + w1);
      int wp = mode ? (w2 * 16 + gy) : (gy * 7 + w2);
      const float* xp = xin + (size_t)b * 128 * HW + hh * 112 + wp;
      float v0 = xp[(size_t)(lane     ) * HW];
      float v1 = xp[(size_t)(lane + 32) * HW];
      float v2 = xp[(size_t)(lane + 64) * HW];
      float v3 = xp[(size_t)(lane + 96) * HW];
      float sum = v0 + v1 + v2 + v3;
      for (int o = 16; o; o >>= 1) sum += __shfl_xor(sum, o, 32);
      float mean = sum * (1.0f / 128.0f);
      float d0 = v0 - mean, d1 = v1 - mean, d2 = v2 - mean, d3 = v3 - mean;
      float vs = d0 * d0 + d1 * d1 + d2 * d2 + d3 * d3;
      for (int o = 16; o; o >>= 1) vs += __shfl_xor(vs, o, 32);
      float inv = rsqrtf(vs * (1.0f / 128.0f) + 1e-5f);
      LNX[t * 136 + lane     ] = (h16)(d0 * inv * ln_g[lane     ] + ln_b[lane     ]);
      LNX[t * 136 + lane + 32] = (h16)(d1 * inv * ln_g[lane + 32] + ln_b[lane + 32]);
      LNX[t * 136 + lane + 64] = (h16)(d2 * inv * ln_g[lane + 64] + ln_b[lane + 64]);
      LNX[t * 136 + lane + 96] = (h16)(d3 * inv * ln_g[lane + 96] + ln_b[lane + 96]);
    } else {
      for (int d = lane; d < 136; d += 32) LNX[t * 136 + d] = (h16)0.f;
    }
  }
  __syncthreads();

  // Phase 2: qkv GEMM  M=64(t) N=384 K=128; scatter into Q/K (t-major) and VT (d-major)
  for (int job = wave; job < 96; job += 8){
    int mtile = job & 3, ntile = job >> 2;
    v8f acc = {};
    const h16* bBase = qkvT + (size_t)(ntile * 16) * 128;
    #pragma unroll
    for (int kb = 0; kb < 4; ++kb){
      v16h a  = frag_a(&LNX[mtile * 16 * 136 + kb * 32], 136, lane);
      v16h bf = frag_b(bBase + kb * 32, 128, lane);
      acc = wmma16(a, bf, acc);
    }
    int ncol = ntile * 16 + (lane & 15);
    int gsel = lane >> 4;
    #pragma unroll
    for (int r = 0; r < 8; ++r){
      int t = mtile * 16 + r + 8 * gsel;
      float v = acc[r];
      if      (ncol < 128) Q [t * 136 + ncol]        = (h16)(v * 0.17677669529663689f);
      else if (ncol < 256) Kk[t * 136 + (ncol - 128)] = (h16)v;
      else                 VT[(ncol - 256) * 72 + t]  = (h16)v;
    }
  }
  __syncthreads();

  // Phase 3: per head: sim GEMM -> bias+softmax -> AV GEMM
  for (int head = 0; head < 4; ++head){
    for (int job = wave; job < 16; job += 8){     // sim: M=64 N=64 K=32
      int mtile = job & 3, ntile = job >> 2;
      v8f acc = {};
      v16h a  = frag_a(&Q [mtile * 16 * 136 + head * 32], 136, lane);
      v16h bf = frag_b(&Kk[ntile * 16 * 136 + head * 32], 136, lane);
      acc = wmma16(a, bf, acc);
      int j = ntile * 16 + (lane & 15);
      int gsel = lane >> 4;
      #pragma unroll
      for (int r = 0; r < 8; ++r) SIM[(mtile * 16 + r + 8 * gsel) * 68 + j] = acc[r];
    }
    __syncthreads();
    for (int i = wave; i < 64; i += 8){           // softmax rows
      if (i < 49){
        int ix = i / 7, iy = i - ix * 7;
        int j0 = lane, j1 = lane + 32;
        float v0, v1 = -1e30f;
        { int jx = j0 / 7, jy = j0 - jx * 7;
          v0 = SIM[i * 68 + j0] + rel[((ix - jx + 6) * 13 + (iy - jy + 6)) * 4 + head]; }
        if (j1 < 49){
          int jx = j1 / 7, jy = j1 - jx * 7;
          v1 = SIM[i * 68 + j1] + rel[((ix - jx + 6) * 13 + (iy - jy + 6)) * 4 + head]; }
        float mx = fmaxf(v0, v1);
        for (int o = 16; o; o >>= 1) mx = fmaxf(mx, __shfl_xor(mx, o, 32));
        float e0 = __expf(v0 - mx);
        float e1 = (j1 < 49) ? __expf(v1 - mx) : 0.f;
        float sm = e0 + e1;
        for (int o = 16; o; o >>= 1) sm += __shfl_xor(sm, o, 32);
        float rinv = 1.0f / sm;
        ATT[i * 72 + j0] = (h16)(e0 * rinv);
        ATT[i * 72 + j1] = (h16)(e1 * rinv);
      } else {
        ATT[i * 72 + lane] = (h16)0.f;
        ATT[i * 72 + lane + 32] = (h16)0.f;
      }
    }
    __syncthreads();
    for (int job = wave; job < 8; job += 8){      // AV: M=64 N=32 K=64
      int mtile = job & 3, ntile = job >> 2;
      v8f acc = {};
      #pragma unroll
      for (int kb = 0; kb < 2; ++kb){
        v16h a  = frag_a(&ATT[mtile * 16 * 72 + kb * 32], 72, lane);
        v16h bf = frag_b(&VT[(head * 32 + ntile * 16) * 72 + kb * 32], 72, lane);
        acc = wmma16(a, bf, acc);
      }
      int dl = head * 32 + ntile * 16 + (lane & 15);
      int gsel = lane >> 4;
      #pragma unroll
      for (int r = 0; r < 8; ++r) OUT[(mtile * 16 + r + 8 * gsel) * 136 + dl] = (h16)acc[r];
    }
    __syncthreads();
  }

  // Phase 4: out-projection GEMM + residual, scatter to canonical bdhw
  for (int job = wave; job < 32; job += 8){
    int mtile = job & 3, ntile = job >> 2;
    v8f acc = {};
    #pragma unroll
    for (int kb = 0; kb < 4; ++kb){
      v16h a  = frag_a(&OUT[mtile * 16 * 136 + kb * 32], 136, lane);
      v16h bf = frag_b(outT + (size_t)(ntile * 16) * 128 + kb * 32, 128, lane);
      acc = wmma16(a, bf, acc);
    }
    int d = ntile * 16 + (lane & 15);
    int gsel = lane >> 4;
    #pragma unroll
    for (int r = 0; r < 8; ++r){
      int t = mtile * 16 + r + 8 * gsel;
      if (t < 49){
        int w1 = t / 7, w2 = t - w1 * 7;
        int hh = mode ? (w1 * 16 + gx) : (gx * 7 + w1);
        int wp = mode ? (w2 * 16 + gy) : (gy * 7 + w2);
        size_t idx = ((size_t)(b * 128 + d)) * HW + hh * 112 + wp;
        xout[idx] = acc[r] + xin[idx];
      }
    }
  }
}

// ---- K6: FFN (LN -> x@w1+b1 -> GELU -> @w2+b2 -> +residual), 64 tokens/block -
__global__ void k_ffn(const float* __restrict__ xin, float* __restrict__ xout,
                      const float* __restrict__ ln_g, const float* __restrict__ ln_b,
                      const h16* __restrict__ w1T, const float* __restrict__ b1,
                      const h16* __restrict__ w2T, const float* __restrict__ b2){
  extern __shared__ h16 lds[];
  h16* LNX = lds;                 // [64][136]
  h16* Hh  = LNX + 64 * 136;      // [64][520]
  const int HW = 12544;
  int b = blockIdx.y;
  int p0 = blockIdx.x * 64;
  int tid = threadIdx.x, wave = tid >> 5, lane = tid & 31;

  for (int t = wave; t < 64; t += 8){
    const float* xp = xin + (size_t)b * 128 * HW + p0 + t;
    float v0 = xp[(size_t)(lane     ) * HW];
    float v1 = xp[(size_t)(lane + 32) * HW];
    float v2 = xp[(size_t)(lane + 64) * HW];
    float v3 = xp[(size_t)(lane + 96) * HW];
    float sum = v0 + v1 + v2 + v3;
    for (int o = 16; o; o >>= 1) sum += __shfl_xor(sum, o, 32);
    float mean = sum * (1.0f / 128.0f);
    float d0 = v0 - mean, d1 = v1 - mean, d2 = v2 - mean, d3 = v3 - mean;
    float vs = d0 * d0 + d1 * d1 + d2 * d2 + d3 * d3;
    for (int o = 16; o; o >>= 1) vs += __shfl_xor(vs, o, 32);
    float inv = rsqrtf(vs * (1.0f / 128.0f) + 1e-5f);
    LNX[t * 136 + lane     ] = (h16)(d0 * inv * ln_g[lane     ] + ln_b[lane     ]);
    LNX[t * 136 + lane + 32] = (h16)(d1 * inv * ln_g[lane + 32] + ln_b[lane + 32]);
    LNX[t * 136 + lane + 64] = (h16)(d2 * inv * ln_g[lane + 64] + ln_b[lane + 64]);
    LNX[t * 136 + lane + 96] = (h16)(d3 * inv * ln_g[lane + 96] + ln_b[lane + 96]);
  }
  __syncthreads();

  for (int job = wave; job < 128; job += 8){      // GEMM1: M=64 N=512 K=128
    int mtile = job & 3, ntile = job >> 2;
    v8f acc = {};
    #pragma unroll
    for (int kb = 0; kb < 4; ++kb){
      v16h a  = frag_a(&LNX[mtile * 16 * 136 + kb * 32], 136, lane);
      v16h bf = frag_b(w1T + (size_t)(ntile * 16) * 128 + kb * 32, 128, lane);
      acc = wmma16(a, bf, acc);
    }
    int n = ntile * 16 + (lane & 15);
    int gsel = lane >> 4;
    #pragma unroll
    for (int r = 0; r < 8; ++r)
      Hh[(mtile * 16 + r + 8 * gsel) * 520 + n] = (h16)gelu_exact(acc[r] + b1[n]);
  }
  __syncthreads();

  for (int job = wave; job < 32; job += 8){       // GEMM2: M=64 N=128 K=512
    int mtile = job & 3, ntile = job >> 2;
    v8f acc = {};
    #pragma unroll
    for (int kb = 0; kb < 16; ++kb){
      v16h a  = frag_a(&Hh[mtile * 16 * 520 + kb * 32], 520, lane);
      v16h bf = frag_b(w2T + (size_t)(ntile * 16) * 512 + kb * 32, 512, lane);
      acc = wmma16(a, bf, acc);
    }
    int d = ntile * 16 + (lane & 15);
    int gsel = lane >> 4;
    #pragma unroll
    for (int r = 0; r < 8; ++r){
      int t = mtile * 16 + r + 8 * gsel;
      size_t idx = ((size_t)(b * 128 + d)) * HW + p0 + t;
      xout[idx] = acc[r] + b2[d] + xin[idx];
    }
  }
}

// -----------------------------------------------------------------------------
extern "C" void kernel_launch(void* const* d_in, const int* in_sizes, int n_in,
                              void* d_out, int out_size, void* d_ws, size_t ws_size,
                              hipStream_t stream) {
  (void)in_sizes; (void)n_in; (void)out_size; (void)ws_size;
  const float* x       = (const float*)d_in[0];
  const float* conv1_w = (const float*)d_in[1];
  const float* conv1_b = (const float*)d_in[2];
  const float* bn1_g   = (const float*)d_in[3];
  const float* bn1_b   = (const float*)d_in[4];
  const float* dw_w    = (const float*)d_in[5];
  const float* dw_b    = (const float*)d_in[6];
  const float* bn2_g   = (const float*)d_in[7];
  const float* bn2_b   = (const float*)d_in[8];
  const float* se_w1   = (const float*)d_in[9];
  const float* se_w2   = (const float*)d_in[10];
  const float* conv2_w = (const float*)d_in[11];
  const float* conv2_b = (const float*)d_in[12];
  const float* bn3_g   = (const float*)d_in[13];
  const float* bn3_b   = (const float*)d_in[14];

  char* ws = (char*)d_ws;
  // f16 transposed weights pool
  h16* conv1_wT = (h16*)ws;                       // [512][64]
  h16* conv2_wT = conv1_wT + 512 * 64;            // [128][512]
  h16* qkvT[2]; h16* outT[2]; h16* w1T[2]; h16* w2T[2];
  h16* wp = conv2_wT + 128 * 512;
  for (int s = 0; s < 2; ++s){
    qkvT[s] = wp; wp += 384 * 128;                // [384][128]
    outT[s] = wp; wp += 128 * 128;                // [128][128]
    w1T [s] = wp; wp += 512 * 128;                // [512][128]
    w2T [s] = wp; wp += 128 * 512;                // [128][512]
  }
  const size_t off = 1u << 20;
  h16* h1 = (h16*)(ws + off);                                     // 205,520,896 B
  h16* h2 = (h16*)(ws + off + 205520896ull);                      //  51,380,224 B
  float* sums   = (float*)(ws + off + 205520896ull + 51380224ull);
  float* sscale = sums + 2048;
  // x0..x3 overlay the (dead after K2) h1 region
  const size_t XP = 25694208ull;
  float* x0 = (float*)(ws + off);
  float* x1 = (float*)(ws + off + XP);
  float* x2 = (float*)(ws + off + 2 * XP);
  float* x3 = (float*)(ws + off + 3 * XP);
  float* xo = (float*)d_out;

  // weight conversion/transposition
  k_cvt_t<<<(64 * 512 + 255) / 256, 256, 0, stream>>>(conv1_w, conv1_wT, 64, 512);
  k_cvt_t<<<(512 * 128 + 255) / 256, 256, 0, stream>>>(conv2_w, conv2_wT, 512, 128);
  for (int s = 0; s < 2; ++s){
    int base = 15 + 11 * s;
    k_cvt_t<<<(128 * 384 + 255) / 256, 256, 0, stream>>>((const float*)d_in[base + 2], qkvT[s], 128, 384);
    k_cvt_t<<<(128 * 128 + 255) / 256, 256, 0, stream>>>((const float*)d_in[base + 3], outT[s], 128, 128);
    k_cvt_t<<<(128 * 512 + 255) / 256, 256, 0, stream>>>((const float*)d_in[base + 7], w1T[s], 128, 512);
    k_cvt_t<<<(512 * 128 + 255) / 256, 256, 0, stream>>>((const float*)d_in[base + 9], w2T[s], 512, 128);
  }

  // MBConv
  k_conv1 <<<dim3(392, 4, 4),  256, 0, stream>>>(x, conv1_wT, conv1_b, bn1_g, bn1_b, h1);
  k_dwconv<<<dim3(49, 512, 4), 256, 0, stream>>>(h1, dw_w, dw_b, bn2_g, bn2_b, h2);
  k_sesum <<<dim3(512, 4),     256, 0, stream>>>(h2, sums);
  k_se    <<<4,                512, 0, stream>>>(sums, se_w1, se_w2, sscale);
  k_conv2 <<<dim3(98, 1, 4),   256, 0, stream>>>(h2, sscale, conv2_wT, conv2_b, bn3_g, bn3_b, x0);

  const size_t ATTN_LDS = 114688;  // 112 KB (<< 320 KB/WGP)
  const size_t FFN_LDS  = 83968;   //  82 KB

  // Stage a1: block attention + FFN
  k_attn<<<dim3(16, 16, 4), 256, ATTN_LDS, stream>>>(x0, x1,
      (const float*)d_in[15], (const float*)d_in[16], qkvT[0], outT[0],
      (const float*)d_in[19], 0);
  k_ffn <<<dim3(196, 4),    256, FFN_LDS,  stream>>>(x1, x2,
      (const float*)d_in[20], (const float*)d_in[21], w1T[0],
      (const float*)d_in[23], w2T[0], (const float*)d_in[25]);

  // Stage a2: grid attention + FFN
  k_attn<<<dim3(16, 16, 4), 256, ATTN_LDS, stream>>>(x2, x3,
      (const float*)d_in[26], (const float*)d_in[27], qkvT[1], outT[1],
      (const float*)d_in[30], 1);
  k_ffn <<<dim3(196, 4),    256, FFN_LDS,  stream>>>(x3, xo,
      (const float*)d_in[31], (const float*)d_in[32], w1T[1],
      (const float*)d_in[34], w2T[1], (const float*)d_in[36]);
}
